// GATrVolumeModel_25005299597333
// MI455X (gfx1250) — matrix operand order repaired and checked by reading, not compiled, not run on hardware
//
#include <hip/hip_runtime.h>
#include <hip/hip_bf16.h>

typedef __attribute__((ext_vector_type(16))) _Float16 v16h;
typedef __attribute__((ext_vector_type(8)))  float    v8f;

#define NTOK   2048
#define NF     256      // 16 channels * 16 blades
#define NLAYER 10
#define NHEAD  8
#define NKT    (NTOK/16)   // 128 key tiles

// ===================== constexpr G(3,0,1) tables =====================
namespace ga {
constexpr int MASK[16] = {0x0, 0x1,0x2,0x4,0x8, 0x3,0x5,0x9,0x6,0xA,0xC,
                          0x7,0xB,0xD,0xE, 0xF};
constexpr int popc(int v){ int c=0; for(int i=0;i<4;++i) c += (v>>i)&1; return c; }
constexpr int idx_of(int m){ for(int i=0;i<16;++i) if(MASK[i]==m) return i; return 0; }
// sign of concatenating sorted blade a then sorted blade b (count inversions)
constexpr int merge_sign(int a, int b){
  int sw = 0;
  for(int i=0;i<4;++i) if((b>>i)&1) sw += popc(a >> (i+1));
  return (sw & 1) ? -1 : 1;
}
} // namespace ga

struct GATables {
  signed char gpi[16][16], gps[16][16];   // geometric product (e0^2 = 0)
  signed char jni[16][16], jns[16][16];   // equivariant join (dual-wedge-dual)
  signed char grade[16], e0in[16], strip[16], inner[16];
  signed char ib[8];                      // INNER_IDX (blades without e0)
};

constexpr GATables make_tables(){
  GATables T{};
  int dt[16] = {}, dsg[16] = {};
  for(int a=0;a<16;++a){
    int ma = ga::MASK[a];
    T.grade[a] = (signed char)ga::popc(ma);
    T.e0in[a]  = (signed char)((ma & 1) ? 1 : 0);
    T.strip[a] = (signed char)((ma & 1) ? ga::idx_of(ma & ~1) : 0);
    T.inner[a] = (signed char)((ma & 1) ? 0 : 1);
    int c = 0xF ^ ma;
    dt[a]  = ga::idx_of(c);
    dsg[a] = ga::merge_sign(ma, c);
  }
  { int n = 0; for(int a=0;a<16;++a) if(T.inner[a]) T.ib[n++] = (signed char)a; }
  for(int i=0;i<16;++i) for(int j=0;j<16;++j){
    int mi = ga::MASK[i], mj = ga::MASK[j];
    if (mi & mj & 1) { T.gpi[i][j]=0; T.gps[i][j]=0; }
    else { T.gpi[i][j]=(signed char)ga::idx_of(mi^mj);
           T.gps[i][j]=(signed char)ga::merge_sign(mi,mj); }
    int ci = dt[i], cj = dt[j];
    int mci = ga::MASK[ci], mcj = ga::MASK[cj];
    if (mci & mcj) { T.jni[i][j]=0; T.jns[i][j]=0; }
    else {
      int sw = ga::merge_sign(mci, mcj);
      int w  = ga::idx_of(mci ^ mcj);
      int r = 0, sr = 1;
      for(int a2=0;a2<16;++a2) if(dt[a2]==w){ r=a2; sr=dsg[a2]; }
      T.jni[i][j] = (signed char)r;
      T.jns[i][j] = (signed char)(dsg[i]*dsg[j]*sw*sr);
    }
  }
  return T;
}
__constant__ GATables TBL = make_tables();

union AFrag { v16h h; unsigned u[8]; _Float16 e[16]; };

// Wfull element value (dense 256x256 equivariant-linear matrix)
__device__ inline float wfull_val(const float* W, int fin, int fout){
  int i = fin >> 4, kb = fin & 15;
  int o = fout >> 4, j = fout & 15;
  float v = 0.f;
  if (kb == j)                           v += W[(o*16+i)*9 + TBL.grade[j]];
  if (TBL.e0in[j] && kb == TBL.strip[j]) v += W[(o*16+i)*9 + 4 + TBL.grade[j]];
  return v;
}

// ===================== kernels =====================

// Build all 70 equi-linear matrices DIRECTLY in WMMA B-fragment register order:
// dword d = ((nt*8 + kc)*32 + lane)*8 + v holds halfs (k0, k0+1) of column nt*16+n,
// where n=lane&15, g=lane>>4, k0 = kc*32 + 16*g + 2*v.
__global__ void build_wfull_k(const float* __restrict__ wq, const float* __restrict__ wk,
                              const float* __restrict__ wv, const float* __restrict__ wo,
                              const float* __restrict__ wl, const float* __restrict__ wr,
                              const float* __restrict__ wm, _Float16* __restrict__ wf){
  int b = blockIdx.x;            // 0..69
  int layer = b / 7, wt = b % 7;
  const float* W = nullptr;
  switch (wt) { case 0: W=wq; break; case 1: W=wk; break; case 2: W=wv; break;
                case 3: W=wo; break; case 4: W=wl; break; case 5: W=wr; break;
                default: W=wm; break; }
  W += layer * 16*16*9;
  _Float16* out = wf + (size_t)b * NF * NF;
  for (int it = 0; it < 128; ++it) {
    int d = it*256 + threadIdx.x;          // dword index 0..32767
    int v    = d & 7;
    int lane = (d >> 3) & 31;
    int kc   = (d >> 8) & 7;
    int nt   = d >> 11;
    int n = lane & 15, g = lane >> 4;
    int k0  = kc*32 + 16*g + 2*v;
    int col = nt*16 + n;
    out[2*(size_t)d]     = (_Float16)wfull_val(W, k0,   col);
    out[2*(size_t)d + 1] = (_Float16)wfull_val(W, k0+1, col);
  }
}

// embed points + W_in -> x[NTOK, 256] f32.  grid 128, block 256 (16 tok x 16 out-ch)
__global__ void embed_k(const float* __restrict__ pts, const float* __restrict__ Win,
                        float* __restrict__ x){
  int t = blockIdx.x*16 + (threadIdx.x >> 4);
  int o = threadIdx.x & 15;
  float px = pts[t*3+0], py = pts[t*3+1], pz = pts[t*3+2];
  float w3 = Win[o*9+3], w8 = Win[o*9+8];
  float* dst = x + (size_t)t*NF + o*16;
  #pragma unroll
  for (int j = 0; j < 16; ++j) dst[j] = 0.f;
  dst[11] = w3*pz; dst[12] = -w3*py; dst[13] = w3*px; dst[14] = w3; dst[15] = w8;
}

// equi_norm: grid NTOK, block 256 -> f16 normalized output
__global__ void norm_k(const float* __restrict__ x, _Float16* __restrict__ xn){
  __shared__ float red[256];
  int t = blockIdx.x, f = threadIdx.x, j = f & 15;
  float v = x[(size_t)t*NF + f];
  red[f] = TBL.inner[j] ? v*v : 0.f;
  __syncthreads();
  for (int s = 128; s > 0; s >>= 1) { if (f < s) red[f] += red[f+s]; __syncthreads(); }
  float factor = red[0] * (1.0f/16.0f);
  float r = rsqrtf(factor + 1e-6f);
  xn[(size_t)t*NF + f] = (_Float16)(v*r);
}

// Y = A(f16 [2048,256] row-major) x B(fragment-swizzled f16 256x256)
// Y16 != null -> f16 out, else Y32 += result.  grid (16, 128), block 32 (one wave)
__global__ void gemm_k(const _Float16* __restrict__ A, const _Float16* __restrict__ Bsw,
                       _Float16* __restrict__ Y16, float* __restrict__ Y32){
  int lane = threadIdx.x;
  int m = lane & 15, g = lane >> 4;
  int col0 = blockIdx.x * 16, row0 = blockIdx.y * 16;
  v8f acc = {};
  const _Float16* arow = A + (size_t)(row0 + m) * NF;
  const unsigned* bw = (const unsigned*)Bsw + ((size_t)blockIdx.x*8*32 + lane)*8;
  for (int kc = 0; kc < 8; ++kc) {
    AFrag a, b;
    #pragma unroll
    for (int v = 0; v < 8; ++v) {
      int kb = 16*g + 2*v + kc*32;     // A 16x32 layout
      a.u[v] = *(const unsigned*)(arow + kb);
    }
    #pragma unroll
    for (int v = 0; v < 8; ++v) b.u[v] = bw[kc*32*8 + v];      // pre-swizzled B
    acc = __builtin_amdgcn_wmma_f32_16x16x32_f16(false, a.h, false, b.h,
                                                 (short)0, acc, false, false);
  }
  #pragma unroll
  for (int gv = 0; gv < 8; ++gv) {
    int row = row0 + gv + (g ? 8 : 0);
    int col = col0 + m;
    if (Y16) Y16[(size_t)row*NF + col] = (_Float16)acc[gv];
    else     Y32[(size_t)row*NF + col] += acc[gv];
  }
}

// pack q/k inner-blade features per head and transpose v per key tile:
// qp,kp: [8][2048][16]   vt: [8][128][32 feat][16 key]   grid 128, block 256
__global__ void qkvpack_k(const _Float16* __restrict__ qh, const _Float16* __restrict__ kh,
                          const _Float16* __restrict__ vh, _Float16* __restrict__ qp,
                          _Float16* __restrict__ kp, _Float16* __restrict__ vt){
  int t = blockIdx.x*16 + (threadIdx.x >> 4);
  int s = threadIdx.x & 15;
  int fs = (s >> 3)*16 + TBL.ib[s & 7];      // within-head feature offset
  #pragma unroll
  for (int h = 0; h < NHEAD; ++h) {
    int feat = 2*h*16 + fs;
    qp[((size_t)h*NTOK + t)*16 + s] = qh[(size_t)t*NF + feat];
    kp[((size_t)h*NTOK + t)*16 + s] = kh[(size_t)t*NF + feat];
    #pragma unroll
    for (int cc = 0; cc < 2; ++cc) {
      vt[(((size_t)h*NKT + (t >> 4))*32 + cc*16 + s)*16 + (t & 15)]
        = vh[(size_t)t*NF + (2*h+cc)*16 + s];
    }
  }
}

// flash attention, one wave per (head, 16-query tile). grid (8, 128), block 32
__global__ void attn_k(const _Float16* __restrict__ qp, const _Float16* __restrict__ kp,
                       const _Float16* __restrict__ vt, _Float16* __restrict__ out){
  __shared__ _Float16 lds[16*16];
  int h = blockIdx.x, qt = blockIdx.y;
  int lane = threadIdx.x, m = lane & 15, g = lane >> 4;

  // Q fragment: A[16 tok x 32]; real K=0..15, zero-pad 16..31 (contiguous dwords)
  AFrag qf;
  const _Float16* qrow = qp + ((size_t)h*NTOK + qt*16 + m)*16;
  #pragma unroll
  for (int v = 0; v < 8; ++v)
    qf.u[v] = (g == 0) ? *(const unsigned*)(qrow + 2*v) : 0u;

  float m_r[8], l_r[8];
  #pragma unroll
  for (int i = 0; i < 8; ++i) { m_r[i] = -1e30f; l_r[i] = 0.f; }
  v8f o0 = {}, o1 = {};

  for (int kt = 0; kt < NKT; ++kt) {
    const _Float16* krow  = kp + ((size_t)h*NTOK + kt*16 + m)*16;
    const _Float16* v0row = vt + (((size_t)h*NKT + kt)*32 + m)*16;
    if (kt + 1 < NKT) {                       // prefetch next K/V tile rows
      __builtin_prefetch(krow + 256, 0, 1);
      __builtin_prefetch(v0row + 512, 0, 1);
    }
    // K fragment: lanes 0-15 carry real K=0..15 (one token's 16 features, 8 dwords)
    AFrag kb;
    #pragma unroll
    for (int v = 0; v < 8; ++v)
      kb.u[v] = (g == 0) ? *(const unsigned*)(krow + 2*v) : 0u;

    v8f zero = {};
    v8f s = __builtin_amdgcn_wmma_f32_16x16x32_f16(false, qf.h, false, kb.h,
                                                   (short)0, zero, false, false);
    #pragma unroll
    for (int gv = 0; gv < 8; ++gv) {
      float sv = s[gv] * 0.25f;               // 1/sqrt(8*ch) = 1/4
      float rmax = sv;
      for (int off = 1; off < 16; off <<= 1)
        rmax = fmaxf(rmax, __shfl_xor(rmax, off));
      float mn = fmaxf(m_r[gv], rmax);
      float p  = __expf(sv - mn);
      float alpha = __expf(m_r[gv] - mn);
      float rsum = p;
      for (int off = 1; off < 16; off <<= 1) rsum += __shfl_xor(rsum, off);
      l_r[gv] = l_r[gv]*alpha + rsum;
      m_r[gv] = mn;
      o0[gv] *= alpha; o1[gv] *= alpha;
      int row = gv + (g ? 8 : 0);
      lds[row*16 + m] = (_Float16)p;          // stage P (C/D layout -> LDS)
    }
    __syncthreads();
    AFrag pf;                                  // reload P as A fragment (K pad to 32)
    #pragma unroll
    for (int v = 0; v < 8; ++v) {
      #pragma unroll
      for (int half = 0; half < 2; ++half) {
        int k = 16*g + 2*v + half;
        pf.e[2*v + half] = (k < 16) ? lds[m*16 + k] : (_Float16)0.f;
      }
    }
    // V fragments from transposed tile: feature-major rows, keys contiguous
    AFrag vb0, vb1;
    #pragma unroll
    for (int v = 0; v < 8; ++v) {
      vb0.u[v] = (g == 0) ? *(const unsigned*)(v0row + 2*v)       : 0u;
      vb1.u[v] = (g == 0) ? *(const unsigned*)(v0row + 256 + 2*v) : 0u;  // +16 feats
    }
    o0 = __builtin_amdgcn_wmma_f32_16x16x32_f16(false, pf.h, false, vb0.h,
                                                (short)0, o0, false, false);
    o1 = __builtin_amdgcn_wmma_f32_16x16x32_f16(false, pf.h, false, vb1.h,
                                                (short)0, o1, false, false);
    __syncthreads();
  }
  #pragma unroll
  for (int gv = 0; gv < 8; ++gv) {
    float inv = 1.0f / l_r[gv];
    int tok = qt*16 + gv + (g ? 8 : 0);
    out[(size_t)tok*NF + (2*h+0)*16 + m] = (_Float16)(o0[gv]*inv);
    out[(size_t)tok*NF + (2*h+1)*16 + m] = (_Float16)(o1[gv]*inv);
  }
}

// geometric product (ch 0..7) / join (ch 8..15) + scalar-gated GELU -> f16
__global__ void gpjoin_k(const _Float16* __restrict__ L, const _Float16* __restrict__ R,
                         _Float16* __restrict__ H){
  int t = blockIdx.x*16 + (threadIdx.x >> 4);
  int c = threadIdx.x & 15;
  float xl[16], xr[16], z[16];
  const _Float16* lp = L + (size_t)t*NF + c*16;
  const _Float16* rp = R + (size_t)t*NF + c*16;
  #pragma unroll
  for (int j = 0; j < 16; ++j) { xl[j] = (float)lp[j]; xr[j] = (float)rp[j]; z[j] = 0.f; }
  if (c < 8) {
    for (int i = 0; i < 16; ++i)
      for (int j = 0; j < 16; ++j) {
        int s = TBL.gps[i][j];
        if (s) z[TBL.gpi[i][j]] += (float)s * xl[i] * xr[j];
      }
  } else {
    for (int i = 0; i < 16; ++i)
      for (int j = 0; j < 16; ++j) {
        int s = TBL.jns[i][j];
        if (s) z[TBL.jni[i][j]] += (float)s * xl[i] * xr[j];
      }
  }
  float g0 = z[0];
  float gate = 0.5f * g0 * (1.0f + erff(g0 * 0.70710678f));
  _Float16* hp = H + (size_t)t*NF + c*16;
  #pragma unroll
  for (int j = 0; j < 16; ++j) hp[j] = (_Float16)(z[j] * gate);
}

// final: mean over tokens of (W_out . scalar blade) -> d_out[0]
__global__ void final_k(const float* __restrict__ x, const float* __restrict__ Wout,
                        float* __restrict__ out){
  __shared__ float red[256];
  float s = 0.f;
  for (int t = threadIdx.x; t < NTOK; t += 256) {
    float acc = 0.f;
    #pragma unroll
    for (int i = 0; i < 16; ++i) acc += Wout[i*9] * x[(size_t)t*NF + i*16];
    s += acc;
  }
  red[threadIdx.x] = s;
  __syncthreads();
  for (int st = 128; st > 0; st >>= 1) { if (threadIdx.x < st) red[threadIdx.x] += red[threadIdx.x+st]; __syncthreads(); }
  if (threadIdx.x == 0) out[0] = red[0] * (1.0f / (float)NTOK);
}

// ===================== host orchestration =====================
extern "C" void kernel_launch(void* const* d_in, const int* in_sizes, int n_in,
                              void* d_out, int out_size, void* d_ws, size_t ws_size,
                              hipStream_t stream) {
  (void)in_sizes; (void)n_in; (void)out_size; (void)ws_size;
  const float* points = (const float*)d_in[0];
  const float* W_in   = (const float*)d_in[1];
  const float* Wq     = (const float*)d_in[2];
  const float* Wk     = (const float*)d_in[3];
  const float* Wv     = (const float*)d_in[4];
  const float* Wo     = (const float*)d_in[5];
  const float* Wl     = (const float*)d_in[6];
  const float* Wr     = (const float*)d_in[7];
  const float* Wm     = (const float*)d_in[8];
  const float* W_out  = (const float*)d_in[9];

  char* ws = (char*)d_ws;
  const size_t F32B = (size_t)NTOK*NF*4;   // 2 MB
  const size_t F16B = (size_t)NTOK*NF*2;   // 1 MB
  size_t off = 0;
  float*    x  = (float*)   (ws + off); off += F32B;
  _Float16* xn = (_Float16*)(ws + off); off += F16B;
  _Float16* qh = (_Float16*)(ws + off); off += F16B;
  _Float16* kh = (_Float16*)(ws + off); off += F16B;
  _Float16* vh = (_Float16*)(ws + off); off += F16B;
  _Float16* ao = (_Float16*)(ws + off); off += F16B;
  _Float16* lh = (_Float16*)(ws + off); off += F16B;
  _Float16* rh = (_Float16*)(ws + off); off += F16B;
  _Float16* hh = (_Float16*)(ws + off); off += F16B;
  _Float16* qp = (_Float16*)(ws + off); off += (size_t)NHEAD*NTOK*16*2;   // 512 KB
  _Float16* kp = (_Float16*)(ws + off); off += (size_t)NHEAD*NTOK*16*2;   // 512 KB
  _Float16* vt = (_Float16*)(ws + off); off += (size_t)NHEAD*NKT*32*16*2; // 1 MB
  _Float16* wf = (_Float16*)(ws + off);                                   // 70 * 128 KB

  dim3 gemmGrid(NF/16, NTOK/16);        // (16, 128)
  dim3 attnGrid(NHEAD, NTOK/16);        // (8, 128)

  build_wfull_k<<<NLAYER*7, 256, 0, stream>>>(Wq, Wk, Wv, Wo, Wl, Wr, Wm, wf);
  embed_k<<<NTOK/16, 256, 0, stream>>>(points, W_in, x);

  for (int layer = 0; layer < NLAYER; ++layer) {
    _Float16* wfl = wf + (size_t)layer*7*NF*NF;
    // --- attention block ---
    norm_k<<<NTOK, 256, 0, stream>>>(x, xn);
    gemm_k<<<gemmGrid, 32, 0, stream>>>(xn, wfl + 0*NF*NF, qh, nullptr);
    gemm_k<<<gemmGrid, 32, 0, stream>>>(xn, wfl + 1*NF*NF, kh, nullptr);
    gemm_k<<<gemmGrid, 32, 0, stream>>>(xn, wfl + 2*NF*NF, vh, nullptr);
    qkvpack_k<<<NTOK/16, 256, 0, stream>>>(qh, kh, vh, qp, kp, vt);
    attn_k<<<attnGrid, 32, 0, stream>>>(qp, kp, vt, ao);
    gemm_k<<<gemmGrid, 32, 0, stream>>>(ao, wfl + 3*NF*NF, nullptr, x);   // x += Wo.ao
    // --- geometric MLP block ---
    norm_k<<<NTOK, 256, 0, stream>>>(x, xn);
    gemm_k<<<gemmGrid, 32, 0, stream>>>(xn, wfl + 4*NF*NF, lh, nullptr);
    gemm_k<<<gemmGrid, 32, 0, stream>>>(xn, wfl + 5*NF*NF, rh, nullptr);
    gpjoin_k<<<NTOK/16, 256, 0, stream>>>(lh, rh, hh);
    gemm_k<<<gemmGrid, 32, 0, stream>>>(hh, wfl + 6*NF*NF, nullptr, x);   // x += Wm.h
  }
  final_k<<<1, 256, 0, stream>>>(x, W_out, (float*)d_out);
}
